// RITS_25958782337187
// MI455X (gfx1250) — matrix operand order repaired and checked by reading, not compiled, not run on hardware
//
#include <hip/hip_runtime.h>

// ---------------------------------------------------------------------------
// RITS persistent-recurrence kernel for gfx1250 (MI455X, wave32, WMMA bf16)
//
// B=1024, T=512, D=128, H=256, C=2.
// Batch rows are independent through time -> 64 workgroups x 16 batch rows,
// each workgroup runs the full T-step scan with h,c resident in LDS.
// All GEMMs use v_wmma_f32_16x16x32_bf16 (f32 accumulate).
// A-operands are kept as bf16 shadow copies in LDS (converted once at write),
// B-operands stream from a bf16 weight pool in d_ws (hot in the 192MB L2).
// ---------------------------------------------------------------------------

#define B_  1024
#define T_  512
#define D_  128
#define H_  256
#define C_  2
#define BT  16     // batch rows per workgroup
#define NW  8      // waves per workgroup
#define TPB 256    // threads per block (8 x wave32)

typedef __attribute__((ext_vector_type(16))) __bf16 v16bf;
typedef __attribute__((ext_vector_type(8)))  __bf16 v8bf;
typedef __attribute__((ext_vector_type(8)))  float  v8f;

// bf16 weight-pool layout in workspace (elements)
#define OFF_WDH 0                         // [H][D]   = 32768
#define OFF_WDX (OFF_WDH + H_*D_)         // [D][D]   = 16384
#define OFF_WH  (OFF_WDX + D_*D_)         // [D][H]   = 32768
#define OFF_WF  (OFF_WH  + D_*H_)         // [D][D]   = 16384
#define OFF_WC  (OFF_WF  + D_*D_)         // [D][2D]  = 32768
#define OFF_WG  (OFF_WC  + D_*2*D_)       // [4H][2D+H] = 524288
#define WTOT    (OFF_WG  + 4*H_*(2*D_+H_))  // 655360 elems -> 1,310,720 B
#define BG_BYTE_OFF ((size_t)WTOT * 2)      // f32 combined gate bias [4H]

// LDS layout: f32 region then bf16 region
#define SM_X   0                  // [BT][D] f32
#define SM_M   (SM_X  + BT*D_)    // [BT][D] f32
#define SM_XH  (SM_M  + BT*D_)    // [BT][D] f32
#define SM_AL  (SM_XH + BT*D_)    // [BT][D] f32
#define SM_H   (SM_AL + BT*D_)    // [BT][H] f32
#define SM_C   (SM_H  + BT*H_)    // [BT][H] f32
#define SM_G   (SM_C  + BT*H_)    // [BT][4H] f32
#define SM_F32_TOTAL (SM_G + BT*4*H_)          // 32768 floats

#define BM_D   0                  // [BT][D] bf16 (deltas)
#define BM_M   (BM_D  + BT*D_)    // [BT][D] bf16 (mask)
#define BM_GX  (BM_M  + BT*D_)    // [BT][D] bf16 (gamma_x)
#define BM_XC  (BM_GX + BT*D_)    // [BT][D] bf16 (x_c)
#define BM_CC  (BM_XC + BT*D_)    // [BT][D] bf16 (c_c)
#define BM_H   (BM_CC + BT*D_)    // [BT][H] bf16 (h)
#define BM_TOTAL (BM_H + BT*H_)                 // 14336 bf16

#define SMEM_BYTES (SM_F32_TOTAL * 4 + BM_TOTAL * 2)   // 159,744 B

// A fragment (16x32 bf16) straight from a bf16 LDS tile: two ds_load_b128.
// Lane l: row = l%16; lanes 0-15 hold K {kb+0..15},
//         lanes 16-31 hold K {kb+16..31}.
static __device__ __forceinline__ v16bf load_a(const __bf16* src, int ld, int kb) {
  const int lane = threadIdx.x & 31;
  const __bf16* p = src + (lane & 15) * ld + kb + 16 * (lane >> 4);
  v8bf lo = *(const v8bf*)p;
  v8bf hi = *(const v8bf*)(p + 8);
  return __builtin_shufflevector(lo, hi, 0, 1, 2, 3, 4, 5, 6, 7,
                                 8, 9, 10, 11, 12, 13, 14, 15);
}

// B fragment (32x16 bf16) from row-major bf16 weights W[out][in] (ld = in-dim).
// Lane l holds weight row n_base + l%16, K-range kb + 16*(l/16) .. +16 (contig).
static __device__ __forceinline__ v16bf load_b(const __bf16* W, int ld,
                                               int n_base, int kb) {
  const int lane = threadIdx.x & 31;
  const __bf16* p = W + (size_t)(n_base + (lane & 15)) * ld
                      + kb + 16 * (lane >> 4);
  return *(const v16bf*)p;
}

static __device__ __forceinline__ v8f wmma_bf16(v16bf a, v16bf b, v8f c) {
  return __builtin_amdgcn_wmma_f32_16x16x32_bf16(false, a, false, b,
                                                 (short)0, c, false, false);
}

static __device__ __forceinline__ float sigmoidf_(float x) {
  return 1.0f / (1.0f + __expf(-x));
}

// ---------------------------------------------------------------------------
// Kernel 0: convert weights to bf16 pool; build Wg = [Wih | Whh] and bg.
// ---------------------------------------------------------------------------
__global__ void rits_prep_kernel(const float* __restrict__ Wd_h,
                                 const float* __restrict__ Wd_x,
                                 const float* __restrict__ Wh,
                                 const float* __restrict__ Wf,
                                 const float* __restrict__ Wc,
                                 const float* __restrict__ Wih,
                                 const float* __restrict__ Whh,
                                 const float* __restrict__ bih,
                                 const float* __restrict__ bhh,
                                 __bf16* __restrict__ wbf,
                                 float* __restrict__ bg) {
  for (int i = blockIdx.x * blockDim.x + threadIdx.x; i < WTOT;
       i += gridDim.x * blockDim.x) {
    float v;
    if      (i < OFF_WDX) v = Wd_h[i - OFF_WDH];
    else if (i < OFF_WH ) v = Wd_x[i - OFF_WDX];
    else if (i < OFF_WF ) v = Wh  [i - OFF_WH ];
    else if (i < OFF_WC ) v = Wf  [i - OFF_WF ];
    else if (i < OFF_WG ) v = Wc  [i - OFF_WC ];
    else {
      int j = i - OFF_WG;
      int o = j >> 9;          // / (2D+H) = /512
      int k = j & 511;
      v = (k < 2 * D_) ? Wih[(size_t)o * (2 * D_) + k]
                       : Whh[(size_t)o * H_ + (k - 2 * D_)];
    }
    wbf[i] = (__bf16)v;
  }
  int j = blockIdx.x * blockDim.x + threadIdx.x;
  if (j < 4 * H_) bg[j] = bih[j] + bhh[j];
}

// ---------------------------------------------------------------------------
// Kernel 1: persistent recurrence, one workgroup per 16 batch rows.
// ---------------------------------------------------------------------------
__global__ void __launch_bounds__(TPB, 1)
rits_scan_kernel(const float* __restrict__ values,
                 const int*   __restrict__ masks,
                 const float* __restrict__ deltas,
                 const float* __restrict__ bd_h,
                 const float* __restrict__ bd_x,
                 const float* __restrict__ bh,
                 const float* __restrict__ bfr,
                 const float* __restrict__ bc,
                 const float* __restrict__ Wo,
                 const float* __restrict__ bo,
                 const __bf16* __restrict__ wbf,
                 const float* __restrict__ bg,
                 float* __restrict__ out) {
  extern __shared__ char smem_raw[];
  float*  fb = (float*)smem_raw;
  __bf16* bb = (__bf16*)(smem_raw + SM_F32_TOTAL * 4);

  float* sx  = fb + SM_X;
  float* sm  = fb + SM_M;
  float* sxh = fb + SM_XH;
  float* sal = fb + SM_AL;
  float* shh = fb + SM_H;
  float* shc = fb + SM_C;
  float* sg  = fb + SM_G;

  __bf16* sdb  = bb + BM_D;
  __bf16* smb  = bb + BM_M;
  __bf16* sgxb = bb + BM_GX;
  __bf16* sxcb = bb + BM_XC;
  __bf16* sccb = bb + BM_CC;
  __bf16* shb  = bb + BM_H;

  const __bf16* Wdh = wbf + OFF_WDH;
  const __bf16* Wdx = wbf + OFF_WDX;
  const __bf16* Wh  = wbf + OFF_WH;
  const __bf16* Wf  = wbf + OFF_WF;
  const __bf16* Wc  = wbf + OFF_WC;
  const __bf16* Wg  = wbf + OFF_WG;

  const int tid  = threadIdx.x;
  const int wid  = tid >> 5;
  const int lane = tid & 31;
  const int col  = lane & 15;
  const int half = lane >> 4;
  const int b0   = blockIdx.x * BT;

  // h0 = c0 = 0
  for (int i = tid; i < BT * H_; i += TPB) {
    shh[i] = 0.0f; shc[i] = 0.0f; shb[i] = (__bf16)0.0f;
  }
  __syncthreads();

  for (int t = 0; t < T_; ++t) {
    // ---- P0: stage x, m, d tiles for this step (non-temporal streams) --
    for (int i = tid; i < BT * D_; i += TPB) {
      int r  = i >> 7;           // /D
      int dd = i & (D_ - 1);
      size_t g = ((size_t)(b0 + r) * T_ + t) * D_ + dd;
      float xv = __builtin_nontemporal_load(&values[g]);
      float mv = (float)__builtin_nontemporal_load(&masks[g]);
      float dv = __builtin_nontemporal_load(&deltas[g]);
      sx[i] = xv;
      sm[i] = mv;
      smb[i] = (__bf16)mv;
      sdb[i] = (__bf16)dv;
      if (t + 1 < T_) __builtin_prefetch(&values[g + D_], 0, 1);
    }
    __syncthreads();

    // ---- P1: gamma_h (16 N-tiles, decays h) + gamma_x (8 N-tiles) ------
    for (int tile = wid; tile < 24; tile += NW) {
      if (tile < 16) {
        int nb = tile * 16;
        v8f acc = {};
#pragma unroll
        for (int kb = 0; kb < D_; kb += 32)
          acc = wmma_bf16(load_a(sdb, D_, kb), load_b(Wdh, D_, nb, kb), acc);
        int n = nb + col;
        float bias = bd_h[n];
#pragma unroll
        for (int v = 0; v < 8; ++v) {
          int row = half * 8 + v;
          float gh = __expf(-fmaxf(acc[v] + bias, 0.0f));
          float nh = shh[row * H_ + n] * gh;     // h = h * gamma_h
          shh[row * H_ + n] = nh;
          shb[row * H_ + n] = (__bf16)nh;
        }
      } else {
        int nb = (tile - 16) * 16;
        v8f acc = {};
#pragma unroll
        for (int kb = 0; kb < D_; kb += 32)
          acc = wmma_bf16(load_a(sdb, D_, kb), load_b(Wdx, D_, nb, kb), acc);
        int n = nb + col;
        float bias = bd_x[n];
#pragma unroll
        for (int v = 0; v < 8; ++v) {
          int row = half * 8 + v;
          sgxb[row * D_ + n] = (__bf16)__expf(-fmaxf(acc[v] + bias, 0.0f));
        }
      }
    }
    __syncthreads();

    // ---- P2: x_h = h@Wh.T + bh  |  alpha = [gx,m]@Wc.T + bc ------------
    for (int tile = wid; tile < 16; tile += NW) {
      if (tile < 8) {
        int nb = tile * 16;
        v8f acc = {};
#pragma unroll
        for (int kb = 0; kb < H_; kb += 32)
          acc = wmma_bf16(load_a(shb, H_, kb), load_b(Wh, H_, nb, kb), acc);
        int n = nb + col;
        float bias = bh[n];
#pragma unroll
        for (int v = 0; v < 8; ++v)
          sxh[(half * 8 + v) * D_ + n] = acc[v] + bias;
      } else {
        int nb = (tile - 8) * 16;
        v8f acc = {};
#pragma unroll
        for (int kb = 0; kb < D_; kb += 32)
          acc = wmma_bf16(load_a(sgxb, D_, kb), load_b(Wc, 2 * D_, nb, kb), acc);
#pragma unroll
        for (int kb = 0; kb < D_; kb += 32)
          acc = wmma_bf16(load_a(smb, D_, kb), load_b(Wc, 2 * D_, nb, D_ + kb), acc);
        int n = nb + col;
        float bias = bc[n];
#pragma unroll
        for (int v = 0; v < 8; ++v)
          sal[(half * 8 + v) * D_ + n] = acc[v] + bias;
      }
    }
    __syncthreads();

    // ---- P3: x_c = m*x + (1-m)*x_h  (bf16 A-copy only) -----------------
    for (int i = tid; i < BT * D_; i += TPB) {
      float m = sm[i];
      sxcb[i] = (__bf16)(m * sx[i] + (1.0f - m) * sxh[i]);
    }
    __syncthreads();

    // ---- P4: z_h = x_c@Wf.T + bf; c_h; c_c; emit imputations ----------
    for (int tile = wid; tile < 8; tile += NW) {
      int nb = tile * 16;
      v8f acc = {};
#pragma unroll
      for (int kb = 0; kb < D_; kb += 32)
        acc = wmma_bf16(load_a(sxcb, D_, kb), load_b(Wf, D_, nb, kb), acc);
      int n = nb + col;
      float bias = bfr[n];
#pragma unroll
      for (int v = 0; v < 8; ++v) {
        int row = half * 8 + v;
        float z  = acc[v] + bias;
        float al = sal[row * D_ + n];
        float ch = al * z + (1.0f - al) * sxh[row * D_ + n];
        float m  = sm[row * D_ + n];
        float cc = m * sx[row * D_ + n] + (1.0f - m) * ch;
        sccb[row * D_ + n] = (__bf16)cc;
        __builtin_nontemporal_store(
            cc, &out[(size_t)B_ * C_ + ((size_t)(b0 + row) * T_ + t) * D_ + n]);
      }
    }
    __syncthreads();

    // ---- P5: gates = [c_c, m, h] @ Wg.T + (bih+bhh) --------------------
    // Paired N-tiles: each A fragment feeds two accumulators.
    for (int p = wid; p < 32; p += NW) {
      int nb0 = p * 32, nb1 = nb0 + 16;
      v8f a0 = {}, a1 = {};
#pragma unroll
      for (int kb = 0; kb < D_; kb += 32) {
        v16bf a = load_a(sccb, D_, kb);
        a0 = wmma_bf16(a, load_b(Wg, 512, nb0, kb), a0);
        a1 = wmma_bf16(a, load_b(Wg, 512, nb1, kb), a1);
      }
#pragma unroll
      for (int kb = 0; kb < D_; kb += 32) {
        v16bf a = load_a(smb, D_, kb);
        a0 = wmma_bf16(a, load_b(Wg, 512, nb0, D_ + kb), a0);
        a1 = wmma_bf16(a, load_b(Wg, 512, nb1, D_ + kb), a1);
      }
#pragma unroll
      for (int kb = 0; kb < H_; kb += 32) {
        v16bf a = load_a(shb, H_, kb);
        a0 = wmma_bf16(a, load_b(Wg, 512, nb0, 2 * D_ + kb), a0);
        a1 = wmma_bf16(a, load_b(Wg, 512, nb1, 2 * D_ + kb), a1);
      }
      int n0 = nb0 + col, n1 = nb1 + col;
      float bias0 = bg[n0], bias1 = bg[n1];
#pragma unroll
      for (int v = 0; v < 8; ++v) {
        int row = half * 8 + v;
        sg[row * (4 * H_) + n0] = a0[v] + bias0;
        sg[row * (4 * H_) + n1] = a1[v] + bias1;
      }
    }
    __syncthreads();

    // ---- P6: LSTM cell update -----------------------------------------
    for (int i = tid; i < BT * H_; i += TPB) {
      int row = i >> 8;          // /H
      int j   = i & (H_ - 1);
      const float* gr = sg + row * (4 * H_);
      float ii = sigmoidf_(gr[j]);
      float ff = sigmoidf_(gr[H_ + j]);
      float gg = tanhf(gr[2 * H_ + j]);
      float oo = sigmoidf_(gr[3 * H_ + j]);
      float c  = ff * shc[i] + ii * gg;
      shc[i] = c;
      float h = oo * tanhf(c);
      shh[i] = h;
      shb[i] = (__bf16)h;
    }
    __syncthreads();
  }

  // ---- epilogue: y_h = h @ Wo.T + bo  (16 rows x 2 classes) ------------
  if (tid < BT * C_) {
    int row = tid / C_, cls = tid % C_;
    float acc = bo[cls];
    for (int k = 0; k < H_; ++k)
      acc += shh[row * H_ + k] * Wo[cls * H_ + k];
    out[(size_t)(b0 + row) * C_ + cls] = acc;
  }
}

// ---------------------------------------------------------------------------
extern "C" void kernel_launch(void* const* d_in, const int* in_sizes, int n_in,
                              void* d_out, int out_size, void* d_ws, size_t ws_size,
                              hipStream_t stream) {
  const float* values = (const float*)d_in[0];
  const int*   masks  = (const int*)  d_in[1];
  const float* deltas = (const float*)d_in[2];
  const float* Wd_h   = (const float*)d_in[3];
  const float* bd_h   = (const float*)d_in[4];
  const float* Wd_x   = (const float*)d_in[5];
  const float* bd_x   = (const float*)d_in[6];
  const float* Wh     = (const float*)d_in[7];
  const float* bh     = (const float*)d_in[8];
  const float* Wf     = (const float*)d_in[9];
  const float* bfr    = (const float*)d_in[10];
  const float* Wc     = (const float*)d_in[11];
  const float* bc     = (const float*)d_in[12];
  const float* Wih    = (const float*)d_in[13];
  const float* bih    = (const float*)d_in[14];
  const float* Whh    = (const float*)d_in[15];
  const float* bhh    = (const float*)d_in[16];
  const float* Wo     = (const float*)d_in[17];
  const float* bo     = (const float*)d_in[18];

  __bf16* wbf = (__bf16*)d_ws;
  float* bg = (float*)((char*)d_ws + BG_BYTE_OFF);
  float* out = (float*)d_out;

  // Kernel 0: bf16 weight pool + fused gate weight/bias.
  rits_prep_kernel<<<(WTOT + TPB - 1) / TPB, TPB, 0, stream>>>(
      Wd_h, Wd_x, Wh, Wf, Wc, Wih, Whh, bih, bhh, wbf, bg);

  // Kernel 1: persistent recurrence. 64 WGs x 256 threads, ~156 KB LDS each.
  rits_scan_kernel<<<B_ / BT, TPB, SMEM_BYTES, stream>>>(
      values, masks, deltas, bd_h, bd_x, bh, bfr, bc, Wo, bo, wbf, bg, out);
}
